// _GNN_S1_37623913513353
// MI455X (gfx1250) — compile-verified, matrix operand order checked by reading, not run on hardware
//
#include <hip/hip_runtime.h>
#include <hip/hip_bf16.h>

// ---------------------------------------------------------------------------
// GNN layer: s = x x^T (per batch) -> top-k adjacency -> A = D^-1/2 adj D^-1/2
//            agg = A (xV^T + Vb) + (xU^T + Ub) -> BatchNorm over (B,C) per node
//            out = relu(x + y)
// B=1024, N=64, C=512.  bf16 WMMA (f32 accumulate) for all GEMMs.
// gfx1250 paths: v_wmma_f32_16x16x32_bf16, global_load_async_to_lds_b128,
//                s_wait_asynccnt, ds_add_f32, global_prefetch_b8.
// ---------------------------------------------------------------------------

typedef __attribute__((ext_vector_type(16))) __bf16 v16bf;
typedef __attribute__((ext_vector_type(8)))  float  v8f;

#define BATCH 1024
#define NN    64
#define CC    512
#define EPSV  1e-5f

static __device__ __forceinline__ __bf16 f2bf(float f) {
    unsigned u = __builtin_bit_cast(unsigned, f);
    unsigned r = u + 0x7FFFu + ((u >> 16) & 1u);          // round-to-nearest-even
    unsigned short h = (unsigned short)(r >> 16);
    return __builtin_bit_cast(__bf16, h);
}

// K index inside a 16x32 bf16 A/B fragment: element e (0..15), half = lane>>4
static __device__ __forceinline__ int kmap(int e, int half) {
    return e + (half << 3) + ((e >= 8) ? 8 : 0);
}

// ---------------------------------------------------------------------------
// K0: convert fp32 weights to bf16 in workspace; zero BN accumulators.
// ---------------------------------------------------------------------------
__global__ void gnn_convert_weights(const float* __restrict__ U,
                                    const float* __restrict__ V,
                                    __bf16* __restrict__ Ubf,
                                    __bf16* __restrict__ Vbf,
                                    float* __restrict__ stats) {
    int i = blockIdx.x * 256 + threadIdx.x;
    if (blockIdx.x == 0 && threadIdx.x < 128) stats[threadIdx.x] = 0.0f; // sum + sumsq
    if (i < CC * CC) {
        Ubf[i] = f2bf(U[i]);
        Vbf[i] = f2bf(V[i]);
    }
}

// ---------------------------------------------------------------------------
// K1: fully fused per-batch kernel. 1 block (8 wave32) per batch.
// Dynamic LDS layout (bytes, dynamic LDS starts at offset 0):
//   [0      ,  65536) xb  : x_b as bf16              64x512 (row-major)
//   [65536  , 131072) vbT : Vx_b^T as bf16           512x64 (channel-major)
//                           (also fp32 async staging buffer for x chunks)
//   [131072 , 147456) sA  : similarity f32           64x64
//   [147456 , 155648) aA  : normalized adj bf16      64x64
//   [155648 , 156672) thr[64], dinv[64], lsum[64], lsq[64]
// ---------------------------------------------------------------------------
#define SMEM_BYTES   156672
#define STAGE_OFF    65536u   // byte offset of the staging / vbT region

__global__ void __launch_bounds__(256)
gnn_fused(const float* __restrict__ x,
          const __bf16* __restrict__ Ubf,
          const __bf16* __restrict__ Vbf,
          const float* __restrict__ Ub,
          const float* __restrict__ Vb,
          const int* __restrict__ kptr,
          float* __restrict__ agg,
          float* __restrict__ stats) {        // stats[0:64)=sum, [64:128)=sumsq
    extern __shared__ char smem[];
    __bf16* xb   = (__bf16*)smem;
    __bf16* vbT  = (__bf16*)(smem + STAGE_OFF);
    float*  stage = (float*)(smem + STAGE_OFF);
    float*  sA   = (float* )(smem + 131072);
    __bf16* aA   = (__bf16*)(smem + 147456);
    float*  thr  = (float* )(smem + 155648);
    float*  dnv  = thr + 64;
    float*  lsum = dnv + 64;
    float*  lsq  = lsum + 64;

    const int b    = blockIdx.x;
    const int tid  = threadIdx.x;
    const int lane = tid & 31;
    const int wave = tid >> 5;
    const int half = lane >> 4;
    const int l16  = lane & 15;

    if (tid < 64) { lsum[tid] = 0.0f; lsq[tid] = 0.0f; }

    // ---- Phase 1: async-DMA x_b (fp32) through LDS staging, convert to bf16
    const float* xg = x + (size_t)b * (NN * CC);
    for (int chunk = 0; chunk < 2; ++chunk) {
        const float* src = xg + chunk * (32 * CC);
        // 32 rows * 512 f32 = 64 KB = 4096 x 16B; 256 lanes -> 16 iters
        for (int i4 = tid; i4 < (32 * CC) / 4; i4 += 256) {
            const unsigned dst = STAGE_OFF + (unsigned)i4 * 16u;  // LDS byte addr
            const float* g = src + i4 * 4;
            asm volatile("global_load_async_to_lds_b128 %0, %1, off"
                         :: "v"(dst), "v"(g) : "memory");
        }
        asm volatile("s_wait_asynccnt 0x0" ::: "memory");
        __syncthreads();
        for (int i = tid; i < 32 * CC; i += 256)
            xb[chunk * 32 * CC + i] = f2bf(stage[i]);
        __syncthreads();
    }

    // ---- Phase 2: s = xb xb^T  (16 tiles of 16x16, K=512) ----
    for (int t = wave; t < 16; t += 8) {
        const int m0 = (t >> 2) << 4;
        const int n0 = (t & 3) << 4;
        v8f acc = {};
        for (int kb = 0; kb < CC; kb += 32) {
            v16bf a, bm;
#pragma unroll
            for (int e = 0; e < 16; ++e) {
                const int k = kb + kmap(e, half);
                a[e]  = xb[(m0 + l16) * CC + k];
                bm[e] = xb[(n0 + l16) * CC + k];
            }
            acc = __builtin_amdgcn_wmma_f32_16x16x32_bf16(
                false, a, false, bm, (short)0, acc, false, false);
        }
#pragma unroll
        for (int r = 0; r < 8; ++r)
            sA[(m0 + r + half * 8) * NN + n0 + l16] = acc[r];
    }
    __syncthreads();

    // ---- Phase 3: per-row top-k threshold, degree^-1/2 ----
    int knn = kptr[0];
    knn = knn < 1 ? 1 : (knn > 8 ? 8 : knn);
    if (tid < 64) {
        float top[8];
#pragma unroll
        for (int i = 0; i < 8; ++i) top[i] = -3.4e38f;
        const float* row = sA + tid * NN;
        for (int j = 0; j < NN; ++j) {
            float v = row[j];
            if (v > top[knn - 1]) {
                int p = knn - 1;
                while (p > 0 && top[p - 1] < v) { top[p] = top[p - 1]; --p; }
                top[p] = v;
            }
        }
        const float th = top[knn - 1];
        int cnt = 0;
        for (int j = 0; j < NN; ++j) cnt += (row[j] >= th) ? 1 : 0;
        thr[tid] = th;
        dnv[tid] = rsqrtf((float)cnt);
    }
    __syncthreads();
    for (int i = tid; i < NN * NN; i += 256) {
        const int r = i >> 6, c = i & 63;
        const float v = (sA[i] >= thr[r]) ? dnv[r] * dnv[c] : 0.0f;
        aA[i] = f2bf(v);
    }
    __syncthreads();

    // ---- Phase 4a: Vx = xb @ V^T + Vb  -> vbT (bf16, transposed: [c][j]) ----
    for (int t = wave; t < 128; t += 8) {
        const int m0 = (t >> 5) << 4;
        const int n0 = (t & 31) << 4;
        v8f acc = {};
        const unsigned* wrow = (const unsigned*)(Vbf + (size_t)(n0 + l16) * CC);
        __builtin_prefetch(wrow, 0, 0);                 // global_prefetch_b8 (L2-resident)
        for (int kb = 0; kb < CC; kb += 32) {
            v16bf a, bm;
#pragma unroll
            for (int e = 0; e < 16; ++e)
                a[e] = xb[(m0 + l16) * CC + kb + kmap(e, half)];
#pragma unroll
            for (int p = 0; p < 8; ++p) {
                const unsigned u = wrow[(kb >> 1) + p + ((p >= 4) ? 4 : 0) + half * 4];
                bm[2 * p]     = __builtin_bit_cast(__bf16, (unsigned short)(u & 0xffffu));
                bm[2 * p + 1] = __builtin_bit_cast(__bf16, (unsigned short)(u >> 16));
            }
            acc = __builtin_amdgcn_wmma_f32_16x16x32_bf16(
                false, a, false, bm, (short)0, acc, false, false);
        }
        const float bias = Vb[n0 + l16];
        // lane owns channel c = n0+l16, rows m0+half*8 .. +7 -> one b128 store
        __bf16 pk[8];
#pragma unroll
        for (int r = 0; r < 8; ++r) pk[r] = f2bf(acc[r] + bias);
        float4 v4;
        __builtin_memcpy(&v4, pk, 16);
        *(float4*)(&vbT[(n0 + l16) * NN + m0 + half * 8]) = v4;
    }
    __syncthreads();

    // ---- Phase 4b: agg = A @ Vx + xb @ U^T + Ub ; BN partial sums ----
    float* aout = agg + (size_t)b * (NN * CC);
    for (int t = wave; t < 128; t += 8) {
        const int m0 = (t >> 5) << 4;
        const int n0 = (t & 31) << 4;
        const float ubias = Ub[n0 + l16];
        v8f acc = {};
#pragma unroll
        for (int r = 0; r < 8; ++r) acc[r] = ubias;

        // xb @ U^T (K = 512)
        const unsigned* wrow = (const unsigned*)(Ubf + (size_t)(n0 + l16) * CC);
        __builtin_prefetch(wrow, 0, 0);
        for (int kb = 0; kb < CC; kb += 32) {
            v16bf a, bm;
#pragma unroll
            for (int e = 0; e < 16; ++e)
                a[e] = xb[(m0 + l16) * CC + kb + kmap(e, half)];
#pragma unroll
            for (int p = 0; p < 8; ++p) {
                const unsigned u = wrow[(kb >> 1) + p + ((p >= 4) ? 4 : 0) + half * 4];
                bm[2 * p]     = __builtin_bit_cast(__bf16, (unsigned short)(u & 0xffffu));
                bm[2 * p + 1] = __builtin_bit_cast(__bf16, (unsigned short)(u >> 16));
            }
            acc = __builtin_amdgcn_wmma_f32_16x16x32_bf16(
                false, a, false, bm, (short)0, acc, false, false);
        }
        // A @ Vx (K = 64): B[k][n] = Vx[j=k][c] = vbT[c*64 + k] -> contiguous
        for (int kb = 0; kb < NN; kb += 32) {
            v16bf a, bm;
#pragma unroll
            for (int e = 0; e < 16; ++e) {
                const int k = kb + kmap(e, half);
                a[e]  = aA[(m0 + l16) * NN + k];
                bm[e] = vbT[(n0 + l16) * NN + k];
            }
            acc = __builtin_amdgcn_wmma_f32_16x16x32_bf16(
                false, a, false, bm, (short)0, acc, false, false);
        }
#pragma unroll
        for (int r = 0; r < 8; ++r) {
            const int m = m0 + r + half * 8;
            const float v = acc[r];
            aout[m * CC + n0 + l16] = v;
            atomicAdd(&lsum[m], v);        // ds_add_f32
            atomicAdd(&lsq[m], v * v);
        }
    }
    __syncthreads();
    if (tid < 64) {
        atomicAdd(&stats[tid], lsum[tid]);        // global_atomic_add_f32
        atomicAdd(&stats[64 + tid], lsq[tid]);
    }
}

// ---------------------------------------------------------------------------
// K2: finalize BN: scale[n] = rsqrt(var+eps)*bn_w[n]; shift[n] = bn_b - mean*scale
// ---------------------------------------------------------------------------
__global__ void gnn_bn_finalize(float* __restrict__ stats,
                                const float* __restrict__ bnw,
                                const float* __restrict__ bnb) {
    const int n = threadIdx.x;
    if (n >= 64) return;
    const float inv = 1.0f / ((float)BATCH * (float)CC);
    const float mean = stats[n] * inv;
    const float var  = stats[64 + n] * inv - mean * mean;
    const float sc   = rsqrtf(var + EPSV) * bnw[n];
    stats[128 + n] = sc;
    stats[192 + n] = bnb[n] - mean * sc;
}

// ---------------------------------------------------------------------------
// K3: out = relu(x + agg*scale[n] + shift[n]); float4 vectorized
// ---------------------------------------------------------------------------
__global__ void __launch_bounds__(256)
gnn_epilogue(const float* __restrict__ x,
             const float* __restrict__ agg,
             const float* __restrict__ stats,
             float* __restrict__ out) {
    const size_t i = (size_t)blockIdx.x * 256 + threadIdx.x;  // float4 index
    const int n = (int)((i >> 7) & 63);                        // (4i/512) % 64
    const float sc = stats[128 + n];
    const float sh = stats[192 + n];
    const float4 xv = ((const float4*)x)[i];
    const float4 av = ((const float4*)agg)[i];
    float4 o;
    o.x = fmaxf(fmaf(av.x, sc, xv.x + sh), 0.0f);
    o.y = fmaxf(fmaf(av.y, sc, xv.y + sh), 0.0f);
    o.z = fmaxf(fmaf(av.z, sc, xv.z + sh), 0.0f);
    o.w = fmaxf(fmaf(av.w, sc, xv.w + sh), 0.0f);
    ((float4*)out)[i] = o;
}

// ---------------------------------------------------------------------------
extern "C" void kernel_launch(void* const* d_in, const int* in_sizes, int n_in,
                              void* d_out, int out_size, void* d_ws, size_t ws_size,
                              hipStream_t stream) {
    (void)in_sizes; (void)n_in; (void)out_size; (void)ws_size;
    const float* x   = (const float*)d_in[0];
    const float* U_w = (const float*)d_in[1];
    const float* U_b = (const float*)d_in[2];
    const float* V_w = (const float*)d_in[3];
    const float* V_b = (const float*)d_in[4];
    const float* bnw = (const float*)d_in[5];
    const float* bnb = (const float*)d_in[6];
    const int*   knn = (const int*)d_in[7];
    float* out = (float*)d_out;

    // workspace layout
    const size_t AGG_BYTES = (size_t)BATCH * NN * CC * sizeof(float); // 128 MiB
    const size_t W_BYTES   = (size_t)CC * CC * sizeof(__bf16);        // 512 KiB
    float*  agg   = (float*)d_ws;
    __bf16* Ubf   = (__bf16*)((char*)d_ws + AGG_BYTES);
    __bf16* Vbf   = (__bf16*)((char*)d_ws + AGG_BYTES + W_BYTES);
    float*  stats = (float*)((char*)d_ws + AGG_BYTES + 2 * W_BYTES);  // 256 f32

    // allow >64KB dynamic LDS for the fused kernel (host-side, capture-safe)
    (void)hipFuncSetAttribute((const void*)gnn_fused,
                              hipFuncAttributeMaxDynamicSharedMemorySize,
                              SMEM_BYTES);

    gnn_convert_weights<<<(CC * CC + 255) / 256, 256, 0, stream>>>(
        U_w, V_w, Ubf, Vbf, stats);

    gnn_fused<<<BATCH, 256, SMEM_BYTES, stream>>>(
        x, Ubf, Vbf, U_b, V_b, knn, agg, stats);

    gnn_bn_finalize<<<1, 64, 0, stream>>>(stats, bnw, bnb);

    const int total4 = BATCH * NN * CC / 4;
    gnn_epilogue<<<total4 / 256, 256, 0, stream>>>(x, agg, stats, out);
}
